// EssentialMatixModule_15848429322970
// MI455X (gfx1250) — compile-verified, matrix operand order checked by reading, not run on hardware
//
#include <hip/hip_runtime.h>
#include <math.h>

typedef _Float16 half_t;
typedef __attribute__((ext_vector_type(16))) _Float16 v16h;
typedef __attribute__((ext_vector_type(8)))  float    v8f;

#define DEV static __device__ __forceinline__

// ---------------- problem constants ----------------
constexpr int Bn  = 64;     // batch
constexpr int Cc  = 256;    // channels (LN dim, GEMM K)
constexpr int Hh  = 3;      // heads
constexpr int Nn  = 576;    // tokens (24*24)
constexpr int HD  = 64;     // head dim
constexpr int VP  = 80;     // padded head dim + 6 pos (70 -> 80)
constexpr int F3  = 576;    // 3*FEAT_DIM (qkv out cols)
constexpr int KPROJ = 29400; // proj K (2*3*70*70)
constexpr int KP  = 29440;  // padded proj K (920*32)
constexpr int OUTD = 512;
constexpr float SCALE = 0.125f;        // 64^-0.5

// ws element counts (halves unless noted)
constexpr size_t XLN_SZ   = (size_t)2 * Bn * Nn * Cc;
constexpr size_t QK_SZ    = (size_t)Bn * Hh * Nn * HD;
constexpr size_t VP_SZ    = (size_t)Bn * Hh * Nn * VP;
constexpr size_t T_SZ     = (size_t)2 * Bn * Hh * Nn * VP;
constexpr size_t FEAT_SZ  = (size_t)Bn * KP;
constexpr size_t WQKV_SZ  = (size_t)F3 * Cc;          // 147456
constexpr size_t WPROJ_SZ = (size_t)OUTD * KPROJ;     // 15052800
constexpr size_t STATS_SZ = (size_t)2 * Bn * Hh * 4 * Nn;  // floats

// ---------------- WMMA fragment helpers (wave32) ----------------
DEV int lane_id() { return threadIdx.x & 31; }

// K index pattern for 16-bit A/B fragments (CDNA5 ISA §7.12.2)
DEV int frag_k(int h, int lane) {
  return h + ((h >= 8) ? 8 : 0) + ((lane & 16) ? 8 : 0);
}

// A fragment 16x32 (MxK): f(row, k)
template <typename F>
DEV v16h load_a_frag(F&& f) {
  int lane = lane_id();
  int row  = lane & 15;
  v16h a;
#pragma unroll
  for (int h = 0; h < 16; ++h) a[h] = f(row, frag_k(h, lane));
  return a;
}

// B fragment 32x16 (KxN): f(k, col)   (column striped per lane)
template <typename F>
DEV v16h load_b_frag(F&& f) {
  int lane = lane_id();
  int col  = lane & 15;
  v16h b;
#pragma unroll
  for (int h = 0; h < 16; ++h) b[h] = f(frag_k(h, lane), col);
  return b;
}

DEV v8f wmma16(v16h a, v16h b, v8f c) {
  return __builtin_amdgcn_wmma_f32_16x16x32_f16(
      /*neg_a=*/false, a, /*neg_b=*/false, b,
      /*c_mod=*/(short)0, c, /*reuse_a=*/false, /*reuse_b=*/false);
}

DEV v8f zero8() {
  v8f z;
#pragma unroll
  for (int i = 0; i < 8; ++i) z[i] = 0.f;
  return z;
}

DEV void atomicMaxF(float* addr, float v) {
  int* ai = reinterpret_cast<int*>(addr);
  int cur = __float_as_int(*addr);
  while (v > __int_as_float(cur)) {
    int prev = atomicCAS(ai, cur, __float_as_int(v));
    if (prev == cur) break;
    cur = prev;
  }
}

// async copy 16B global -> LDS (gfx1250; tracked by ASYNCcnt)
DEV void async_g2l_b128(unsigned lds_byte_addr, const void* gaddr) {
  asm volatile("global_load_async_to_lds_b128 %0, %1, off"
               :: "v"(lds_byte_addr), "v"(gaddr) : "memory");
}
DEV void wait_asynccnt0() {
  asm volatile("s_wait_asynccnt 0x0" ::: "memory");
}

// ---------------- 0) fp32 -> fp16 weight conversion ----------------
__global__ void cvt_kernel(const float* __restrict__ src, half_t* __restrict__ dst, size_t n) {
  size_t idx = (size_t)blockIdx.x * blockDim.x + threadIdx.x;
  if (idx < n) dst[idx] = (half_t)src[idx];
}

// ---------------- 1) LayerNorm + transpose -> fp16 ----------------
__global__ void ln_kernel(const float* __restrict__ x1, const float* __restrict__ x2,
                          const float* __restrict__ lnw, const float* __restrict__ lnb,
                          half_t* __restrict__ xln) {
  int wave = (blockIdx.x * blockDim.x + threadIdx.x) >> 5;
  int lane = lane_id();
  if (wave >= 2 * Bn * Nn) return;
  int t = wave / (Bn * Nn);
  int rem = wave % (Bn * Nn);
  int b = rem / Nn, n = rem % Nn;
  const float* x = (t == 0) ? x1 : x2;
  const float* row = x + (size_t)b * Cc * Nn + n;   // stride Nn over channel
  float vals[8], s = 0.f;
#pragma unroll
  for (int i = 0; i < 8; ++i) { vals[i] = row[(size_t)(lane + 32 * i) * Nn]; s += vals[i]; }
#pragma unroll
  for (int off = 16; off > 0; off >>= 1) s += __shfl_xor(s, off, 32);
  float mu = s * (1.f / Cc), vs = 0.f;
#pragma unroll
  for (int i = 0; i < 8; ++i) { float d = vals[i] - mu; vs += d * d; }
#pragma unroll
  for (int off = 16; off > 0; off >>= 1) vs += __shfl_xor(vs, off, 32);
  float rstd = rsqrtf(vs * (1.f / Cc) + 1e-5f);
  half_t* orow = xln + ((size_t)t * Bn * Nn + rem) * Cc;
#pragma unroll
  for (int i = 0; i < 8; ++i) {
    int c = lane + 32 * i;
    orow[c] = (half_t)((vals[i] - mu) * rstd * lnw[c] + lnb[c]);
  }
}

// ---------------- 2) QKV GEMM (WMMA, f16 weights) + scatter ----------------
__global__ void qkv_kernel(const half_t* __restrict__ xln, const half_t* __restrict__ w16,
                           half_t* __restrict__ Q0, half_t* __restrict__ Q1,
                           half_t* __restrict__ K0, half_t* __restrict__ K1,
                           half_t* __restrict__ V0, half_t* __restrict__ V1) {
  const int MT = (Bn * Nn) / 16, NT = F3 / 16;     // 2304, 36
  int wave = (blockIdx.x * blockDim.x + threadIdx.x) >> 5;
  if (wave >= 2 * MT * NT) return;
  int t = wave / (MT * NT);
  int rem = wave % (MT * NT);
  int mt = rem / NT, nt = rem % NT;
  const half_t* A = xln + (size_t)t * Bn * Nn * Cc + (size_t)(mt * 16) * Cc;
  v8f acc = zero8();
#pragma unroll
  for (int kc = 0; kc < Cc / 32; ++kc) {
    __builtin_prefetch(A + (kc + 1) * 32, 0, 0);
    v16h a = load_a_frag([&](int r, int k) { return A[(size_t)r * Cc + kc * 32 + k]; });
    v16h b = load_b_frag([&](int k, int c) {
      return w16[(size_t)(nt * 16 + c) * Cc + kc * 32 + k]; });
    acc = wmma16(a, b, acc);
  }
  int lane = lane_id();
  int colb = nt * 16 + (lane & 15);
  int which = colb / 192, h = (colb % 192) / HD, d = colb % HD;
  int rofs = (lane & 16) ? 8 : 0;
#pragma unroll
  for (int g = 0; g < 8; ++g) {
    int grow = mt * 16 + g + rofs;
    int b = grow / Nn, n = grow % Nn;
    size_t bhn = ((size_t)b * Hh + h) * Nn + n;
    half_t v = (half_t)acc[g];
    if (which == 0)      (t ? Q1 : Q0)[bhn * HD + d] = v;
    else if (which == 1) (t ? K1 : K0)[bhn * HD + d] = v;
    else                 (t ? V1 : V0)[bhn * VP + d] = v;
  }
}

// ---------------- 3) append positional features into V-padded ----------------
__global__ void pos_kernel(half_t* __restrict__ V0, half_t* __restrict__ V1) {
  int idx = blockIdx.x * blockDim.x + threadIdx.x;
  if (idx >= 2 * Bn * Hh * Nn) return;
  int t = idx / (Bn * Hh * Nn);
  int r = idx % (Bn * Hh * Nn);
  int n = r % Nn;
  float p3 = -1.f + 2.f * (float)(n % 24) / 23.f;   // ys tiled
  float p4 = -1.f + 2.f * (float)(n / 24) / 23.f;   // xs repeated
  half_t* v = (t ? V1 : V0) + (size_t)r * VP;
  v[64] = (half_t)(p3 * p3); v[65] = (half_t)(p4 * p4); v[66] = (half_t)(p3 * p4);
  v[67] = (half_t)p3;        v[68] = (half_t)p4;        v[69] = (half_t)1.f;
#pragma unroll
  for (int c = 70; c < VP; ++c) v[c] = (half_t)0.f;
}

// S tile (16x16) at (ti,tj) from Q,K [Nn x HD], scaled
DEV v8f compute_S(const half_t* Q, const half_t* K, int ti, int tj) {
  v8f s = zero8();
#pragma unroll
  for (int kc = 0; kc < HD / 32; ++kc) {
    v16h a = load_a_frag([&](int r, int k) { return Q[(size_t)(ti * 16 + r) * HD + kc * 32 + k]; });
    v16h b = load_b_frag([&](int k, int c) { return K[(size_t)(tj * 16 + c) * HD + kc * 32 + k]; });
    s = wmma16(a, b, s);
  }
#pragma unroll
  for (int g = 0; g < 8; ++g) s[g] *= SCALE;
  return s;
}

// ---------------- 4) dual-softmax row/col stats (2 WMMA passes over S) ----------------
__global__ void stats_kernel(const half_t* __restrict__ Q0, const half_t* __restrict__ Q1,
                             const half_t* __restrict__ K0, const half_t* __restrict__ K1,
                             float* __restrict__ stats) {
  int dir = blockIdx.x / (Bn * Hh);
  int bh  = blockIdx.x % (Bn * Hh);
  const half_t* Q = (dir == 0 ? Q1 : Q0) + (size_t)bh * Nn * HD;
  const half_t* K = (dir == 0 ? K0 : K1) + (size_t)bh * Nn * HD;
  __shared__ float smem[4 * Nn];   // rmax | cmax | rsum | csum
  for (int i = threadIdx.x; i < 2 * Nn; i += blockDim.x) smem[i] = -1e30f;
  for (int i = threadIdx.x; i < 2 * Nn; i += blockDim.x) smem[2 * Nn + i] = 0.f;
  __syncthreads();
  int wave = threadIdx.x >> 5, nw = blockDim.x >> 5, lane = lane_id();
  int col16 = lane & 15, rofs = (lane & 16) ? 8 : 0;
  const int TT = 36 * 36;
  // pass 1: maxes
  for (int tile = wave; tile < TT; tile += nw) {
    int ti = tile / 36, tj = tile % 36;
    v8f s = compute_S(Q, K, ti, tj);
    int m = tj * 16 + col16;
#pragma unroll
    for (int g = 0; g < 8; ++g) {
      int n = ti * 16 + g + rofs;
      atomicMaxF(&smem[n], s[g]);
      atomicMaxF(&smem[Nn + m], s[g]);
    }
  }
  __syncthreads();
  // pass 2: exp-sums
  for (int tile = wave; tile < TT; tile += nw) {
    int ti = tile / 36, tj = tile % 36;
    v8f s = compute_S(Q, K, ti, tj);
    int m = tj * 16 + col16;
#pragma unroll
    for (int g = 0; g < 8; ++g) {
      int n = ti * 16 + g + rofs;
      atomicAdd(&smem[2 * Nn + n], __expf(s[g] - smem[n]));
      atomicAdd(&smem[3 * Nn + m], __expf(s[g] - smem[Nn + m]));
    }
  }
  __syncthreads();
  float* st = stats + ((size_t)dir * (Bn * Hh) + bh) * (4 * Nn);
  for (int i = threadIdx.x; i < 4 * Nn; i += blockDim.x) st[i] = smem[i];
}

// ---- 5) T = dualsoftmax(S) @ Vp  (S recomputed; V tile via async LDS copy) ----
__global__ void av_kernel(const half_t* __restrict__ Q0, const half_t* __restrict__ Q1,
                          const half_t* __restrict__ K0, const half_t* __restrict__ K1,
                          const half_t* __restrict__ V0, const half_t* __restrict__ V1,
                          const float* __restrict__ stats, half_t* __restrict__ T) {
  int blk = blockIdx.x;                      // dir*(BH*9) + bh*9 + nc
  int dir = blk / (Bn * Hh * 9);
  int r   = blk % (Bn * Hh * 9);
  int bh  = r / 9, nc = r % 9;
  const half_t* Q = (dir == 0 ? Q1 : Q0) + (size_t)bh * Nn * HD;
  const half_t* K = (dir == 0 ? K0 : K1) + (size_t)bh * Nn * HD;
  const half_t* V = (dir == 0 ? V0 : V1) + (size_t)bh * Nn * VP;
  const float*  st = stats + ((size_t)dir * (Bn * Hh) + bh) * (4 * Nn);
  __shared__ half_t astage[4][16 * 32];
  __shared__ half_t vstage[32 * VP];         // shared by all 4 waves per m-chunk
  int wave = threadIdx.x >> 5, lane = lane_id();
  int ti = nc * 4 + wave;                    // n-tile 0..35
  int col16 = lane & 15, rofs = (lane & 16) ? 8 : 0;
  unsigned vstage_base = (unsigned)(size_t)&vstage[0];  // LDS byte address (low 32 bits)
  v8f acc[5];
#pragma unroll
  for (int e = 0; e < 5; ++e) acc[e] = zero8();
  for (int mc = 0; mc < Nn / 32; ++mc) {     // 18 m-chunks
    // kick off async copy of the 32x80 V tile into LDS (overlaps S compute)
    const half_t* Vb = V + (size_t)(mc * 32) * VP;
    for (int i = threadIdx.x; i < 320; i += blockDim.x) {   // 320 x 16B chunks
      int row = i / 10, c8 = i % 10;
      async_g2l_b128(vstage_base + (unsigned)(row * VP + c8 * 8) * 2,
                     Vb + (size_t)row * VP + c8 * 8);
    }
    // compute two 16-wide S tiles for this m-chunk while the copy flies
#pragma unroll
    for (int hmf = 0; hmf < 2; ++hmf) {
      int tj = mc * 2 + hmf;
      v8f s = compute_S(Q, K, ti, tj);
      int m = tj * 16 + col16;
      float cm = st[Nn + m], cs = st[3 * Nn + m];
#pragma unroll
      for (int g = 0; g < 8; ++g) {
        int n = ti * 16 + g + rofs;
        float a = __expf(2.f * s[g] - st[n] - cm) / (st[2 * Nn + n] * cs);
        astage[wave][(g + rofs) * 32 + hmf * 16 + col16] = (half_t)a;
      }
    }
    wait_asynccnt0();                        // drain this wave's async copies
    __syncthreads();                         // all waves' copies + astage visible
    v16h afrag = load_a_frag([&](int rr, int kk) { return astage[wave][rr * 32 + kk]; });
#pragma unroll
    for (int eg = 0; eg < 5; ++eg) {
      v16h bfrag = load_b_frag([&](int k, int c) {
        return vstage[(size_t)k * VP + eg * 16 + c]; });
      acc[eg] = wmma16(afrag, bfrag, acc[eg]);
    }
    __syncthreads();
  }
  half_t* To = T + ((size_t)dir * (Bn * Hh) + bh) * Nn * VP;
#pragma unroll
  for (int eg = 0; eg < 5; ++eg)
#pragma unroll
    for (int g = 0; g < 8; ++g) {
      int n = ti * 16 + g + rofs;
      To[(size_t)n * VP + eg * 16 + col16] = (half_t)acc[eg][g];
    }
}

// ---------------- 6) feat = Vpᵀ @ T  (per dir,b,h -> [70,70] block) ----------------
__global__ void zero_feat_kernel(half_t* __restrict__ feat) {
  size_t idx = (size_t)blockIdx.x * blockDim.x + threadIdx.x;
  if (idx < FEAT_SZ) feat[idx] = (half_t)0.f;
}

__global__ void vtav_kernel(const half_t* __restrict__ V0, const half_t* __restrict__ V1,
                            const half_t* __restrict__ Tbuf, half_t* __restrict__ feat) {
  int dir = blockIdx.x / (Bn * Hh);
  int bh  = blockIdx.x % (Bn * Hh);
  int b = bh / Hh, h = bh % Hh;
  const half_t* V = (dir == 0 ? V0 : V1) + (size_t)bh * Nn * VP;
  const half_t* T = Tbuf + ((size_t)dir * (Bn * Hh) + bh) * Nn * VP;
  int wave = threadIdx.x >> 5;               // 25 waves: 5x5 output tiles
  int dt = wave / 5, et = wave % 5;
  int lane = lane_id();
  int col16 = lane & 15, rofs = (lane & 16) ? 8 : 0;
  v8f acc = zero8();
  for (int kc = 0; kc < Nn / 32; ++kc) {     // K = 576
    v16h a = load_a_frag([&](int rr, int k) {   // A[d, n] = V[n, d]  (transposed gather)
      return V[(size_t)(kc * 32 + k) * VP + dt * 16 + rr]; });
    v16h bb = load_b_frag([&](int k, int c) {
      return T[(size_t)(kc * 32 + k) * VP + et * 16 + c]; });
    acc = wmma16(a, bb, acc);
  }
  int e = et * 16 + col16;
#pragma unroll
  for (int g = 0; g < 8; ++g) {
    int d = dt * 16 + g + rofs;
    if (d < 70 && e < 70)
      feat[(size_t)b * KP + (size_t)dir * 14700 + h * 4900 + d * 70 + e] = (half_t)acc[g];
  }
}

// ---------------- 7) out = relu(feat @ projWᵀ + b)  (WMMA, f16 weights) ----------------
__global__ void proj_kernel(const half_t* __restrict__ feat, const half_t* __restrict__ w16,
                            const float* __restrict__ bias, float* __restrict__ out) {
  int wave = (blockIdx.x * blockDim.x + threadIdx.x) >> 5;
  const int MT = Bn / 16, NT = OUTD / 16;    // 4, 32
  if (wave >= MT * NT) return;
  int mt = wave / NT, nt = wave % NT;
  int lane = lane_id();
  int col16 = lane & 15, rofs = (lane & 16) ? 8 : 0;
  v8f acc = zero8();
  for (int kc = 0; kc < KP / 32; ++kc) {     // 920 chunks
    __builtin_prefetch(feat + (size_t)(mt * 16) * KP + (kc + 1) * 32, 0, 0);
    v16h a = load_a_frag([&](int r, int k) {
      return feat[(size_t)(mt * 16 + r) * KP + kc * 32 + k]; });
    v16h bb = load_b_frag([&](int k, int c) {
      int kg = kc * 32 + k;
      return (kg < KPROJ) ? w16[(size_t)(nt * 16 + c) * KPROJ + kg] : (half_t)0.f; });
    acc = wmma16(a, bb, acc);
  }
  int o = nt * 16 + col16;
  float bo = bias[o];
#pragma unroll
  for (int g = 0; g < 8; ++g) {
    int b = mt * 16 + g + rofs;
    float v = acc[g] + bo;
    out[(size_t)b * OUTD + o] = v > 0.f ? v : 0.f;
  }
}

// ---------------- host launch ----------------
extern "C" void kernel_launch(void* const* d_in, const int* in_sizes, int n_in,
                              void* d_out, int out_size, void* d_ws, size_t ws_size,
                              hipStream_t stream) {
  (void)in_sizes; (void)n_in; (void)out_size; (void)ws_size;
  const float* x1   = (const float*)d_in[0];
  const float* x2   = (const float*)d_in[1];
  const float* lnw  = (const float*)d_in[2];
  const float* lnb  = (const float*)d_in[3];
  const float* qkvw = (const float*)d_in[4];
  const float* projw= (const float*)d_in[5];
  const float* projb= (const float*)d_in[6];
  float* out = (float*)d_out;

  char* p = (char*)d_ws;
  half_t* xln  = (half_t*)p; p += XLN_SZ * sizeof(half_t);
  half_t* Q0   = (half_t*)p; p += QK_SZ * sizeof(half_t);
  half_t* Q1   = (half_t*)p; p += QK_SZ * sizeof(half_t);
  half_t* K0   = (half_t*)p; p += QK_SZ * sizeof(half_t);
  half_t* K1   = (half_t*)p; p += QK_SZ * sizeof(half_t);
  half_t* V0   = (half_t*)p; p += VP_SZ * sizeof(half_t);
  half_t* V1   = (half_t*)p; p += VP_SZ * sizeof(half_t);
  half_t* T    = (half_t*)p; p += T_SZ * sizeof(half_t);
  half_t* feat = (half_t*)p; p += FEAT_SZ * sizeof(half_t);
  half_t* wq16 = (half_t*)p; p += WQKV_SZ * sizeof(half_t);
  half_t* wp16 = (half_t*)p; p += WPROJ_SZ * sizeof(half_t);
  float*  stats = (float*)p; p += STATS_SZ * sizeof(float);

  // 0) weight conversions (once per launch)
  cvt_kernel<<<(int)((WQKV_SZ + 255) / 256), 256, 0, stream>>>(qkvw, wq16, WQKV_SZ);
  cvt_kernel<<<(int)((WPROJ_SZ + 255) / 256), 256, 0, stream>>>(projw, wp16, WPROJ_SZ);
  // 1) LN: 2*B*N waves, 8 waves/block
  ln_kernel<<<(2 * Bn * Nn) / 8, 256, 0, stream>>>(x1, x2, lnw, lnb, xln);
  // 2) QKV GEMM: 2*2304*36 tiles, 8 waves/block
  qkv_kernel<<<(2 * 2304 * 36) / 8, 256, 0, stream>>>(xln, wq16, Q0, Q1, K0, K1, V0, V1);
  // 3) positional append
  pos_kernel<<<(2 * Bn * Hh * Nn + 255) / 256, 256, 0, stream>>>(V0, V1);
  // 4) dual-softmax stats: one block per (dir,b,h)
  stats_kernel<<<2 * Bn * Hh, 256, 0, stream>>>(Q0, Q1, K0, K1, stats);
  // 5) T = A @ Vp: block per (dir,b,h,nchunk), 4 waves, async V staging
  av_kernel<<<2 * Bn * Hh * 9, 128, 0, stream>>>(Q0, Q1, K0, K1, V0, V1, stats, T);
  // 6) feat = Vpᵀ @ T
  zero_feat_kernel<<<(int)((FEAT_SZ + 255) / 256), 256, 0, stream>>>(feat);
  vtav_kernel<<<2 * Bn * Hh, 800, 0, stream>>>(V0, V1, T, feat);
  // 7) projection + relu
  proj_kernel<<<16, 256, 0, stream>>>(feat, wp16, projb, out);
}
